// WordSAGEBLOCK2_78159814852866
// MI455X (gfx1250) — matrix operand order repaired and checked by reading, not compiled, not run on hardware
//
#include <hip/hip_runtime.h>
#include <hip/hip_bf16.h>

// ---------------- problem constants ----------------
#define N_GENE  20000
#define N_TRAIN 4096
#define SRCF    2500
#define DSTF    2675
#define HD      2500     // H == OUT == SRC == 2500
#define NC      16
#define TRF     (DSTF - NC)   // 2659
#define EPSV    1e-5f
#define SLOPE   0.01f

typedef __bf16 bf16;
typedef __attribute__((ext_vector_type(4)))  __bf16 v4bf;
typedef __attribute__((ext_vector_type(16))) __bf16 v16bf;
typedef __attribute__((ext_vector_type(8)))  float  v8f;

// ---------------- utility kernels ----------------
__global__ __launch_bounds__(256) void zero_kernel(float* __restrict__ p, long n) {
  long i = (long)blockIdx.x * blockDim.x + threadIdx.x;
  long stride = (long)gridDim.x * blockDim.x;
  for (; i < n; i += stride) p[i] = 0.0f;
}

__global__ __launch_bounds__(256) void deg_kernel(const int* __restrict__ ed,
                                                  float* __restrict__ deg, int ne) {
  int i = blockIdx.x * blockDim.x + threadIdx.x;
  if (i < ne) unsafeAtomicAdd(&deg[ed[i]], 1.0f);
}

// one block per edge: agg[dst] += gene[src]
__global__ __launch_bounds__(256) void scatter_kernel(const float* __restrict__ gene,
                                                      const int* __restrict__ es,
                                                      const int* __restrict__ ed,
                                                      float* __restrict__ agg) {
  int e = blockIdx.x;
  int s = es[e], d = ed[e];
  const float* gs = gene + (long)s * SRCF;
  float*       ag = agg  + (long)d * SRCF;
  for (int c = threadIdx.x; c < SRCF; c += blockDim.x)
    unsafeAtomicAdd(&ag[c], gs[c]);
}

__global__ __launch_bounds__(256) void mean_div_kernel(float* __restrict__ agg,
                                                       const float* __restrict__ deg) {
  long n = (long)N_TRAIN * SRCF;
  long i = (long)blockIdx.x * blockDim.x + threadIdx.x;
  long stride = (long)gridDim.x * blockDim.x;
  for (; i < n; i += stride) {
    float d = deg[i / SRCF];
    agg[i] = agg[i] / fmaxf(d, 1.0f);
  }
}

// per-column batch statistics (thread per column -> coalesced over columns)
__global__ __launch_bounds__(256) void colstats_kernel(const float* __restrict__ X,
                                                       int M, int N,
                                                       float* __restrict__ cm,
                                                       float* __restrict__ crs) {
  int c = blockIdx.x * blockDim.x + threadIdx.x;
  if (c >= N) return;
  float s = 0.f, s2 = 0.f;
  for (int i = 0; i < M; ++i) {
    float v = X[(long)i * N + c];
    s += v; s2 += v * v;
  }
  float m   = s / (float)M;
  float var = s2 / (float)M - m * m;
  cm[c]  = m;
  crs[c] = rsqrtf(var + EPSV);
}

__global__ __launch_bounds__(256) void bn_leaky_kernel(const float* __restrict__ X,
                                                       const float* __restrict__ cm,
                                                       const float* __restrict__ crs,
                                                       const float* __restrict__ g,
                                                       const float* __restrict__ b,
                                                       float* __restrict__ Y,
                                                       int M, int N) {
  long n = (long)M * N;
  long i = (long)blockIdx.x * blockDim.x + threadIdx.x;
  long stride = (long)gridDim.x * blockDim.x;
  for (; i < n; i += stride) {
    int c = (int)(i % N);
    float v = (X[i] - cm[c]) * crs[c] * g[c] + b[c];
    Y[i] = (v >= 0.f) ? v : SLOPE * v;
  }
}

// row softmax, one block per row
__global__ __launch_bounds__(256) void softmax_rows(float* __restrict__ S, int L) {
  float* p = S + (long)blockIdx.x * L;
  __shared__ float red[256];
  int t = threadIdx.x;
  float mx = -3.4e38f;
  for (int c = t; c < L; c += 256) mx = fmaxf(mx, p[c]);
  red[t] = mx; __syncthreads();
  for (int s = 128; s > 0; s >>= 1) { if (t < s) red[t] = fmaxf(red[t], red[t + s]); __syncthreads(); }
  mx = red[0]; __syncthreads();
  float sum = 0.f;
  for (int c = t; c < L; c += 256) { float e = expf(p[c] - mx); p[c] = e; sum += e; }
  red[t] = sum; __syncthreads();
  for (int s = 128; s > 0; s >>= 1) { if (t < s) red[t] += red[t + s]; __syncthreads(); }
  float inv = 1.0f / red[0];
  for (int c = t; c < L; c += 256) p[c] *= inv;
}

// row LayerNorm + leaky, one block per row
__global__ __launch_bounds__(256) void ln_leaky_kernel(const float* __restrict__ X,
                                                       const float* __restrict__ g,
                                                       const float* __restrict__ b,
                                                       float* __restrict__ Y, int N) {
  const float* x = X + (long)blockIdx.x * N;
  float*       y = Y + (long)blockIdx.x * N;
  __shared__ float r1[256], r2[256];
  int t = threadIdx.x;
  float s = 0.f, s2 = 0.f;
  for (int c = t; c < N; c += 256) { float v = x[c]; s += v; s2 += v * v; }
  r1[t] = s; r2[t] = s2; __syncthreads();
  for (int st = 128; st > 0; st >>= 1) {
    if (t < st) { r1[t] += r1[t + st]; r2[t] += r2[t + st]; }
    __syncthreads();
  }
  float m   = r1[0] / (float)N;
  float var = r2[0] / (float)N - m * m;
  float rs  = rsqrtf(var + EPSV);
  for (int c = t; c < N; c += 256) {
    float v = (x[c] - m) * rs * g[c] + b[c];
    y[c] = (v >= 0.f) ? v : SLOPE * v;
  }
}

// ---------------- WMMA GEMM: C = act(alpha*A@B(^T) + beta*C + bias) ----------------
// bf16 hi/lo split (a=ah+al, b=bh+bl; keep ah*bh + ah*bl + al*bh) -> ~fp32 accuracy
// 128x128x32 block tile, 8 waves (4x2), each wave computes 32x64 (8 f32 accumulators)
#define BM 128
#define BN 128
#define BK 32
#define KSTRIDE 40   // padded LDS row stride (bf16 elems): bank-friendly, keeps 8B align for k%4==0

// convert 4 f32 -> packed hi/lo bf16 quads
__device__ __forceinline__ void cvt4(float v0, float v1, float v2, float v3,
                                     v4bf& hv, v4bf& lv) {
  bf16 h0 = (bf16)v0, h1 = (bf16)v1, h2 = (bf16)v2, h3 = (bf16)v3;
  hv = (v4bf){h0, h1, h2, h3};
  lv = (v4bf){(bf16)(v0 - (float)h0), (bf16)(v1 - (float)h1),
              (bf16)(v2 - (float)h2), (bf16)(v3 - (float)h3)};
}

__global__ __launch_bounds__(256) void gemm_wmma_kernel(
    const float* __restrict__ A, int lda,
    const float* __restrict__ B, int ldb, int transB,
    float* __restrict__ C, int ldc,
    int M, int N, int K,
    const float* __restrict__ bias,
    float alpha, float beta, int act) {
  __shared__ __attribute__((aligned(16))) bf16 Ah[BM * KSTRIDE];
  __shared__ __attribute__((aligned(16))) bf16 Al[BM * KSTRIDE];
  __shared__ __attribute__((aligned(16))) bf16 Bh[BN * KSTRIDE];   // transposed: [n][k]
  __shared__ __attribute__((aligned(16))) bf16 Bl[BN * KSTRIDE];

  const int tid  = threadIdx.x;
  const int lane = tid & 31;
  const int wid  = tid >> 5;       // 8 waves
  const int wm   = wid >> 1;       // 0..3 -> 32-row strip
  const int wn   = wid & 1;        // 0..1 -> 64-col strip
  const int half = lane >> 4;
  const int l15  = lane & 15;
  const int bm0  = blockIdx.y * BM;
  const int bn0  = blockIdx.x * BN;

  v8f acc[2][4];
#pragma unroll
  for (int rg = 0; rg < 2; ++rg)
#pragma unroll
    for (int cg = 0; cg < 4; ++cg)
      acc[rg][cg] = (v8f){0.f,0.f,0.f,0.f,0.f,0.f,0.f,0.f};

  for (int k0 = 0; k0 < K; k0 += BK) {
    // ---- stage A tile (128x32) in groups of 4 k-elements ----
#pragma unroll
    for (int g = tid; g < BM * (BK / 4); g += 256) {
      int m  = g >> 3;
      int kk = (g & 7) * 4;
      int gm = bm0 + m, gk = k0 + kk;
      float v0 = 0.f, v1 = 0.f, v2 = 0.f, v3 = 0.f;
      if (gm < M) {
        const float* ap = A + (long)gm * lda + gk;
        if (gk + 4 <= K) { v0 = ap[0]; v1 = ap[1]; v2 = ap[2]; v3 = ap[3]; }
        else {
          if (gk + 0 < K) v0 = ap[0];
          if (gk + 1 < K) v1 = ap[1];
          if (gk + 2 < K) v2 = ap[2];
          if (gk + 3 < K) v3 = ap[3];
        }
      }
      v4bf hv, lv; cvt4(v0, v1, v2, v3, hv, lv);
      *(v4bf*)&Ah[m * KSTRIDE + kk] = hv;
      *(v4bf*)&Al[m * KSTRIDE + kk] = lv;
    }
    // ---- stage B tile transposed to [n][k], groups of 4 k-elements ----
    if (transB) {        // B is [N][K] row-major: contiguous in k -> vector-friendly
#pragma unroll
      for (int g = tid; g < BN * (BK / 4); g += 256) {
        int n  = g >> 3;
        int kk = (g & 7) * 4;
        int gn = bn0 + n, gk = k0 + kk;
        float v0 = 0.f, v1 = 0.f, v2 = 0.f, v3 = 0.f;
        if (gn < N) {
          const float* bp = B + (long)gn * ldb + gk;
          if (gk + 4 <= K) { v0 = bp[0]; v1 = bp[1]; v2 = bp[2]; v3 = bp[3]; }
          else {
            if (gk + 0 < K) v0 = bp[0];
            if (gk + 1 < K) v1 = bp[1];
            if (gk + 2 < K) v2 = bp[2];
            if (gk + 3 < K) v3 = bp[3];
          }
        }
        v4bf hv, lv; cvt4(v0, v1, v2, v3, hv, lv);
        *(v4bf*)&Bh[n * KSTRIDE + kk] = hv;
        *(v4bf*)&Bl[n * KSTRIDE + kk] = lv;
      }
    } else {             // B is [K][N]: coalesce over n, strided k loads per thread
#pragma unroll
      for (int g = tid; g < BN * (BK / 4); g += 256) {
        int n  = g & (BN - 1);
        int kk = (g >> 7) * 4;
        int gn = bn0 + n, gk = k0 + kk;
        float v0 = 0.f, v1 = 0.f, v2 = 0.f, v3 = 0.f;
        if (gn < N) {
          const float* bp = B + (long)gk * ldb + gn;
          if (gk + 4 <= K) {
            v0 = bp[0]; v1 = bp[(long)ldb]; v2 = bp[2L * ldb]; v3 = bp[3L * ldb];
          } else {
            if (gk + 0 < K) v0 = bp[0];
            if (gk + 1 < K) v1 = bp[(long)ldb];
            if (gk + 2 < K) v2 = bp[2L * ldb];
            if (gk + 3 < K) v3 = bp[3L * ldb];
          }
        }
        v4bf hv, lv; cvt4(v0, v1, v2, v3, hv, lv);
        *(v4bf*)&Bh[n * KSTRIDE + kk] = hv;
        *(v4bf*)&Bl[n * KSTRIDE + kk] = lv;
      }
    }
    __syncthreads();

    // ---- A fragments: lane-half K chunks {0-15} / {16-31} ----
    v16bf afh[2], afl[2];
#pragma unroll
    for (int rg = 0; rg < 2; ++rg) {
      const bf16* prh = &Ah[(wm * 32 + rg * 16 + l15) * KSTRIDE];
      const bf16* prl = &Al[(wm * 32 + rg * 16 + l15) * KSTRIDE];
      int b0 = half * 16, b1 = half * 16 + 8;
#pragma unroll
      for (int i = 0; i < 8; ++i) {
        afh[rg][i] = prh[b0 + i]; afh[rg][8 + i] = prh[b1 + i];
        afl[rg][i] = prl[b0 + i]; afl[rg][8 + i] = prl[b1 + i];
      }
    }
    // ---- 4 col-groups x 2 row-groups x 3 split-WMMAs = 24 WMMAs / wave / K-step ----
#pragma unroll
    for (int cg = 0; cg < 4; ++cg) {
      int col = wn * 64 + cg * 16 + l15;
      const bf16* pbh = &Bh[col * KSTRIDE + half * 16];
      const bf16* pbl = &Bl[col * KSTRIDE + half * 16];
      v16bf bfh, bfl;
#pragma unroll
      for (int i = 0; i < 16; ++i) { bfh[i] = pbh[i]; bfl[i] = pbl[i]; }
#pragma unroll
      for (int rg = 0; rg < 2; ++rg) {
        v8f a = acc[rg][cg];
        a = __builtin_amdgcn_wmma_f32_16x16x32_bf16(false, afh[rg], false, bfh, (short)0, a, false, false);
        a = __builtin_amdgcn_wmma_f32_16x16x32_bf16(false, afh[rg], false, bfl, (short)0, a, false, false);
        a = __builtin_amdgcn_wmma_f32_16x16x32_bf16(false, afl[rg], false, bfh, (short)0, a, false, false);
        acc[rg][cg] = a;
      }
    }
    __syncthreads();
  }

  // ---- epilogue: D-layout VGPR r -> row r (+8 for upper lane half) ----
#pragma unroll
  for (int rg = 0; rg < 2; ++rg) {
#pragma unroll
    for (int cg = 0; cg < 4; ++cg) {
#pragma unroll
      for (int r = 0; r < 8; ++r) {
        int row = bm0 + wm * 32 + rg * 16 + r + half * 8;
        int col = bn0 + wn * 64 + cg * 16 + l15;
        if (row < M && col < N) {
          float v = acc[rg][cg][r] * alpha;
          if (beta != 0.0f) v += beta * C[(long)row * ldc + col];
          if (bias) v += bias[col];
          if (act) v = fmaxf(v, 0.0f);
          C[(long)row * ldc + col] = v;
        }
      }
    }
  }
}

// ---------------- host-side orchestration ----------------
static inline void launch_gemm(const float* A, int lda, const float* B, int ldb, int transB,
                               float* C, int ldc, int M, int N, int K,
                               const float* bias, float alpha, float beta, int act,
                               hipStream_t stream) {
  dim3 grid((N + BN - 1) / BN, (M + BM - 1) / BM);
  gemm_wmma_kernel<<<grid, 256, 0, stream>>>(A, lda, B, ldb, transB, C, ldc,
                                             M, N, K, bias, alpha, beta, act);
}

extern "C" void kernel_launch(void* const* d_in, const int* in_sizes, int n_in,
                              void* d_out, int out_size, void* d_ws, size_t ws_size,
                              hipStream_t stream) {
  const float* gene   = (const float*)d_in[0];
  const float* train  = (const float*)d_in[1];
  const float* y_hat  = (const float*)d_in[2];
  const int*   esrc   = (const int*)d_in[3];
  const int*   edst   = (const int*)d_in[4];
  const float* Wself3 = (const float*)d_in[5];
  const float* Wngh3  = (const float*)d_in[6];
  const float* b3     = (const float*)d_in[7];
  const float* bn3g   = (const float*)d_in[8];
  const float* bn3b   = (const float*)d_in[9];
  const float* Wq     = (const float*)d_in[10];
  const float* Wk     = (const float*)d_in[11];
  const float* Wv     = (const float*)d_in[12];
  const float* bq     = (const float*)d_in[13];
  const float* bk     = (const float*)d_in[14];
  const float* bv     = (const float*)d_in[15];
  const float* Wo     = (const float*)d_in[16];
  const float* bo     = (const float*)d_in[17];
  const float* ln2g   = (const float*)d_in[18];
  const float* ln2b   = (const float*)d_in[19];
  const float* Wself4 = (const float*)d_in[20];
  const float* Wngh4  = (const float*)d_in[21];
  const float* b4     = (const float*)d_in[22];
  const float* bn4g   = (const float*)d_in[23];
  const float* bn4b   = (const float*)d_in[24];
  const float* Wlin2  = (const float*)d_in[25];
  const float* blin2  = (const float*)d_in[26];
  const float* Wce    = (const float*)d_in[27];
  const float* bce    = (const float*)d_in[28];
  const int ne = in_sizes[3];

  // workspace layout (floats)
  const size_t SZ = (size_t)N_TRAIN * HD;          // 10,240,000
  float* ws   = (float*)d_ws;
  float* mean = ws;                                // agg then mean (reused conv3+conv4)
  float* h    = ws + 1 * SZ;
  float* qb   = ws + 2 * SZ;
  float* kb   = ws + 3 * SZ;                       // also attn@V temp
  float* vb   = ws + 4 * SZ;                       // also relu(lin2) temp
  float* Sb   = ws + 5 * SZ;                       // 4096*4096 attention scores
  float* deg  = Sb + (size_t)N_TRAIN * N_TRAIN;
  float* cm   = deg + N_TRAIN;
  float* crs  = cm + 2560;

  float* x_hat = (float*)d_out;                    // 4096*2500
  float* y_out = x_hat + SZ;                       // 4096*16

  // ---- graph mean aggregation (shared by conv3/conv4) ----
  zero_kernel<<<20000, 256, 0, stream>>>(mean, (long)SZ);
  zero_kernel<<<16, 256, 0, stream>>>(deg, N_TRAIN);
  deg_kernel<<<(ne + 255) / 256, 256, 0, stream>>>(edst, deg, ne);
  scatter_kernel<<<ne, 256, 0, stream>>>(gene, esrc, edst, mean);
  mean_div_kernel<<<20000, 256, 0, stream>>>(mean, deg);

  // ---- conv3: h = [train|y_hat]@W_self3 + mean@W_neigh3 + b3 ----
  launch_gemm(train, TRF, Wself3, HD, 0, h, HD, N_TRAIN, HD, TRF, b3, 1.f, 0.f, 0, stream);
  launch_gemm(y_hat, NC, Wself3 + (size_t)TRF * HD, HD, 0, h, HD, N_TRAIN, HD, NC, nullptr, 1.f, 1.f, 0, stream);
  launch_gemm(mean, SRCF, Wngh3, HD, 0, h, HD, N_TRAIN, HD, SRCF, nullptr, 1.f, 1.f, 0, stream);
  colstats_kernel<<<(HD + 255) / 256, 256, 0, stream>>>(h, N_TRAIN, HD, cm, crs);
  bn_leaky_kernel<<<20000, 256, 0, stream>>>(h, cm, crs, bn3g, bn3b, h, N_TRAIN, HD);

  // ---- self-attention over all 4096 train nodes ----
  launch_gemm(h, HD, Wq, HD, 0, qb, HD, N_TRAIN, HD, HD, bq, 1.f, 0.f, 0, stream);
  launch_gemm(h, HD, Wk, HD, 0, kb, HD, N_TRAIN, HD, HD, bk, 1.f, 0.f, 0, stream);
  launch_gemm(h, HD, Wv, HD, 0, vb, HD, N_TRAIN, HD, HD, bv, 1.f, 0.f, 0, stream);
  launch_gemm(qb, HD, kb, HD, 1, Sb, N_TRAIN, N_TRAIN, N_TRAIN, HD,
              nullptr, 1.0f / sqrtf((float)HD), 0.f, 0, stream);
  softmax_rows<<<N_TRAIN, 256, 0, stream>>>(Sb, N_TRAIN);
  launch_gemm(Sb, N_TRAIN, vb, HD, 0, kb, HD, N_TRAIN, HD, N_TRAIN, nullptr, 1.f, 0.f, 0, stream);
  launch_gemm(kb, HD, Wo, HD, 0, qb, HD, N_TRAIN, HD, HD, bo, 1.f, 0.f, 0, stream);
  ln_leaky_kernel<<<N_TRAIN, 256, 0, stream>>>(qb, ln2g, ln2b, h, HD);

  // ---- conv4 ----
  launch_gemm(h, HD, Wself4, HD, 0, kb, HD, N_TRAIN, HD, HD, b4, 1.f, 0.f, 0, stream);
  launch_gemm(mean, SRCF, Wngh4, HD, 0, kb, HD, N_TRAIN, HD, SRCF, nullptr, 1.f, 1.f, 0, stream);
  colstats_kernel<<<(HD + 255) / 256, 256, 0, stream>>>(kb, N_TRAIN, HD, cm, crs);
  bn_leaky_kernel<<<20000, 256, 0, stream>>>(kb, cm, crs, bn4g, bn4b, x_hat, N_TRAIN, HD);

  // ---- head: y = relu(x_hat@W_lin2 + b)@W_ce + b_ce ----
  launch_gemm(x_hat, HD, Wlin2, HD, 0, vb, HD, N_TRAIN, HD, HD, blin2, 1.f, 0.f, 1, stream);
  launch_gemm(vb, HD, Wce, NC, 0, y_out, NC, N_TRAIN, NC, HD, bce, 1.f, 0.f, 0, stream);

  (void)n_in; (void)out_size; (void)ws_size;
}
